// GNN_59725815218879
// MI455X (gfx1250) — compile-verified, matrix operand order checked by reading, not run on hardware
//
#include <hip/hip_runtime.h>
#include <hip/hip_bf16.h>
#include <stdint.h>

typedef __bf16 bf16;
typedef __bf16 bf16x16 __attribute__((ext_vector_type(16)));
typedef __bf16 bf16x8  __attribute__((ext_vector_type(8)));
typedef __bf16 bf16x4  __attribute__((ext_vector_type(4)));
typedef float  f32x8   __attribute__((ext_vector_type(8)));

#define NNODES 50000
#define NEDGES 1600000

// ---------------- elementwise helpers ----------------
// convert W (KxN, f32) -> Wt (NxK, bf16): GEMM B-tiles become contiguous loads
__global__ void k_cvt_transpose(const float* __restrict__ W, bf16* __restrict__ Wt,
                                int K, int N) {
    int idx = blockIdx.x * 256 + threadIdx.x;
    if (idx < K * N) {
        int k = idx / N, n = idx % N;
        Wt[(size_t)n * K + k] = (bf16)W[idx];
    }
}

__global__ void k_deg(const int* __restrict__ edges, float* __restrict__ deg, int e_count) {
    int e = blockIdx.x * 256 + threadIdx.x;
    if (e < e_count) atomicAdd(&deg[edges[2 * e + 1]], 1.0f);
}

__global__ void k_dinv(const float* __restrict__ deg, float* __restrict__ dinv, int n) {
    int i = blockIdx.x * 256 + threadIdx.x;
    if (i < n) dinv[i] = rsqrtf(deg[i] + 1.0f);
}

// out[i,f] = h[i,f] * dinv[i]^2   (initializes scatter target: write, not add)
__global__ void k_self(const float* __restrict__ h, const float* __restrict__ dinv,
                       float* __restrict__ out, int n) {
    int idx = blockIdx.x * 256 + threadIdx.x;
    if (idx < n * 64) {
        float di = dinv[idx >> 6];
        out[idx] = h[idx] * di * di;
    }
}

// one wave32 per edge; lane owns features {2*lane, 2*lane+1} (b64 loads)
__global__ void k_scatter(const float* __restrict__ h, const int* __restrict__ edges,
                          const float* __restrict__ dinv, float* __restrict__ out, int e_count) {
    int gid  = blockIdx.x * 256 + threadIdx.x;
    int e    = gid >> 5;
    int lane = threadIdx.x & 31;
    if (e >= e_count) return;
    int2 ed = *(const int2*)(&edges[2 * e]);
    float c = dinv[ed.x] * dinv[ed.y];
    float2 v = *(const float2*)(&h[(size_t)ed.x * 64 + 2 * lane]);
    float* o = &out[(size_t)ed.y * 64 + 2 * lane];
    atomicAdd(o,     v.x * c);
    atomicAdd(o + 1, v.y * c);
}

// dual-buffer scatter: both branches share edges + coefficients -> one edge pass
__global__ void k_scatter2(const float* __restrict__ h0, const float* __restrict__ h1,
                           const int* __restrict__ edges, const float* __restrict__ dinv,
                           float* __restrict__ out0, float* __restrict__ out1, int e_count) {
    int gid  = blockIdx.x * 256 + threadIdx.x;
    int e    = gid >> 5;
    int lane = threadIdx.x & 31;
    if (e >= e_count) return;
    int2 ed = *(const int2*)(&edges[2 * e]);
    float c = dinv[ed.x] * dinv[ed.y];
    size_t so = (size_t)ed.x * 64 + 2 * lane;
    size_t dو = 0; (void)dو;
    float2 a = *(const float2*)(&h0[so]);
    float2 b = *(const float2*)(&h1[so]);
    float* o0 = &out0[(size_t)ed.y * 64 + 2 * lane];
    float* o1 = &out1[(size_t)ed.y * 64 + 2 * lane];
    atomicAdd(o0,     a.x * c);
    atomicAdd(o0 + 1, a.y * c);
    atomicAdd(o1,     b.x * c);
    atomicAdd(o1 + 1, b.y * c);
}

__global__ void k_bias_relu64(float* __restrict__ buf, const float* __restrict__ bias, int n) {
    int idx = blockIdx.x * 256 + threadIdx.x;
    if (idx < n * 64) {
        float v = buf[idx] + bias[idx & 63];
        buf[idx] = v > 0.0f ? v : 0.0f;
    }
}

// ---------------- WMMA bf16 GEMM:  C(MxN) = A(MxK,f32) * Bt(NxK,bf16) ----------------
// block = 256 threads (8 wave32), tile 128x64, K step 32. Software-pipelined:
// global loads of tile k+1 issue before the ds_load/WMMA work on tile k.
#define BM 128
#define BN 64
#define BK 32
#define LDA 40  // bf16 elems per LDS row: 32 + 8 pad (80B; all vector accesses 8/16B aligned)
#define LDB 40

__global__ __launch_bounds__(256)
void k_gemm_wmma(const float* __restrict__ A, const bf16* __restrict__ Bt,
                 const float* __restrict__ bias, float* __restrict__ C,
                 int M, int N, int K, int fuse_bias_relu)
{
    __shared__ __align__(16) bf16 ldsA[BM * LDA];
    __shared__ __align__(16) bf16 ldsB[BN * LDB];   // [n][k]

    const int t    = threadIdx.x;
    const int lane = t & 31;
    const int wave = t >> 5;        // 0..7 -> M sub-tile
    const int half = lane >> 4;     // lane group selects K sub-ranges per ISA layout
    const int l16  = lane & 15;
    const int m0   = blockIdx.x * BM;
    const int n0   = blockIdx.y * BN;

    // ---- per-thread load slots (branch-free, clamped rows) ----
    // A: 128x32 f32 = 1024 float4; thread loads 4 float4 (coalesced b128)
    const int ar[4] = { (t + 0*256) >> 3, (t + 1*256) >> 3, (t + 2*256) >> 3, (t + 3*256) >> 3 };
    const int ac    = (t & 7) * 4;                 // same column group for all 4
    int arow[4];
    #pragma unroll
    for (int i = 0; i < 4; ++i) {
        int row = m0 + ar[i];
        arow[i] = row < M ? row : (M - 1);         // clamp: stores are masked later
    }
    // B: 64 rows x 32 k = 2048 bf16; thread loads one 16B chunk (8 bf16)
    const int bn = t >> 2;                         // 0..63
    const int bc = (t & 3) * 8;                    // 0,8,16,24

    f32x8 acc[4];
    #pragma unroll
    for (int i = 0; i < 4; ++i) acc[i] = f32x8{0.f,0.f,0.f,0.f,0.f,0.f,0.f,0.f};

    float4 aReg[4];
    bf16x8 bReg;

    // prologue: load tile k0 = 0
    #pragma unroll
    for (int i = 0; i < 4; ++i)
        aReg[i] = *(const float4*)(&A[(size_t)arow[i] * K + ac]);
    bReg = *(const bf16x8*)(&Bt[(size_t)(n0 + bn) * K + bc]);

    for (int k0 = 0; k0 < K; k0 += BK) {
        // store current tile regs -> LDS (packed bf16 converts, b64 stores)
        #pragma unroll
        for (int i = 0; i < 4; ++i) {
            bf16x4 b;
            b[0] = (bf16)aReg[i].x; b[1] = (bf16)aReg[i].y;
            b[2] = (bf16)aReg[i].z; b[3] = (bf16)aReg[i].w;
            *(bf16x4*)(&ldsA[ar[i] * LDA + ac]) = b;
        }
        *(bf16x8*)(&ldsB[bn * LDB + bc]) = bReg;
        __syncthreads();

        // issue global loads for next tile (overlap with WMMA below)
        if (k0 + BK < K) {
            #pragma unroll
            for (int i = 0; i < 4; ++i)
                aReg[i] = *(const float4*)(&A[(size_t)arow[i] * K + (k0 + BK) + ac]);
            bReg = *(const bf16x8*)(&Bt[(size_t)(n0 + bn) * K + (k0 + BK) + bc]);
        }

        // A fragment: lane = row l16; K = half*8 + {0..7} and 16 + half*8 + {0..7}
        bf16x16 afrag;
        {
            const bf16* pa = &ldsA[(wave * 16 + l16) * LDA + half * 8];
            bf16x8 lo = *(const bf16x8*)(pa);
            bf16x8 hi = *(const bf16x8*)(pa + 16);
            #pragma unroll
            for (int j = 0; j < 8; ++j) { afrag[j] = lo[j]; afrag[j + 8] = hi[j]; }
        }
        #pragma unroll
        for (int nt = 0; nt < 4; ++nt) {
            bf16x16 bfrag;
            const bf16* pb = &ldsB[(nt * 16 + l16) * LDB + half * 8];
            bf16x8 lo = *(const bf16x8*)(pb);
            bf16x8 hi = *(const bf16x8*)(pb + 16);
            #pragma unroll
            for (int j = 0; j < 8; ++j) { bfrag[j] = lo[j]; bfrag[j + 8] = hi[j]; }
            acc[nt] = __builtin_amdgcn_wmma_f32_16x16x32_bf16(
                false, afrag, false, bfrag, (short)0, acc[nt], false, false);
        }
        __syncthreads();
    }

    // C/D layout: lanes 0-15 -> M = r, lanes 16-31 -> M = 8 + r; N = lane & 15
    #pragma unroll
    for (int nt = 0; nt < 4; ++nt) {
        int col = n0 + nt * 16 + l16;
        #pragma unroll
        for (int r = 0; r < 8; ++r) {
            int row = m0 + wave * 16 + half * 8 + r;
            if (row < M) {
                float v = acc[nt][r];
                if (fuse_bias_relu) { v += bias[col]; v = v > 0.0f ? v : 0.0f; }
                C[(size_t)row * N + col] = v;
            }
        }
    }
}

// ---------------- host orchestration ----------------
static inline int cdiv(int a, int b) { return (a + b - 1) / b; }

extern "C" void kernel_launch(void* const* d_in, const int* in_sizes, int n_in,
                              void* d_out, int out_size, void* d_ws, size_t ws_size,
                              hipStream_t stream)
{
    const float* x     = (const float*)d_in[0];
    const int*   edges = (const int*)  d_in[1];
    const float* Wb    = (const float*)d_in[2];
    const float* bb    = (const float*)d_in[3];
    const float* Ws1   = (const float*)d_in[4];
    const float* bs1   = (const float*)d_in[5];
    const float* Ws2   = (const float*)d_in[6];
    const float* bs2   = (const float*)d_in[7];
    const float* Ww1   = (const float*)d_in[8];
    const float* bw1   = (const float*)d_in[9];
    const float* Ww2   = (const float*)d_in[10];
    const float* bw2   = (const float*)d_in[11];
    float* out = (float*)d_out;

    const int N = NNODES, E = NEDGES;

    // workspace layout (~52 MB)
    char* w = (char*)d_ws;
    float* deg  = (float*)w; w += 50048 * sizeof(float);
    float* dinv = (float*)w; w += 50048 * sizeof(float);
    bf16* Wb_bf  = (bf16*)w; w += 640 * 64 * sizeof(bf16);   // transposed (64 x 640)
    bf16* Ws1_bf = (bf16*)w; w += 64 * 64  * sizeof(bf16);   // (64 x 64)
    bf16* Ws2_bf = (bf16*)w; w += 64 * 640 * sizeof(bf16);   // (640 x 64)
    bf16* Ww1_bf = (bf16*)w; w += 64 * 64  * sizeof(bf16);
    bf16* Ww2_bf = (bf16*)w; w += 64 * 640 * sizeof(bf16);
    w = (char*)(((uintptr_t)w + 255) & ~(uintptr_t)255);
    float* bufA = (float*)w; w += (size_t)N * 64 * sizeof(float);
    float* bufB = (float*)w; w += (size_t)N * 64 * sizeof(float);
    float* bufC = (float*)w; w += (size_t)N * 64 * sizeof(float);
    float* bufD = (float*)w; w += (size_t)N * 64 * sizeof(float);

    hipMemsetAsync(deg, 0, (size_t)N * sizeof(float), stream);

    // bf16 transposed weight copies
    k_cvt_transpose<<<cdiv(640 * 64, 256), 256, 0, stream>>>(Wb,  Wb_bf,  640, 64);
    k_cvt_transpose<<<cdiv(64 * 64, 256),  256, 0, stream>>>(Ws1, Ws1_bf, 64, 64);
    k_cvt_transpose<<<cdiv(64 * 640, 256), 256, 0, stream>>>(Ws2, Ws2_bf, 64, 640);
    k_cvt_transpose<<<cdiv(64 * 64, 256),  256, 0, stream>>>(Ww1, Ww1_bf, 64, 64);
    k_cvt_transpose<<<cdiv(64 * 640, 256), 256, 0, stream>>>(Ww2, Ww2_bf, 64, 640);

    // degree -> dinv
    k_deg <<<cdiv(E, 256), 256, 0, stream>>>(edges, deg, E);
    k_dinv<<<cdiv(N, 256), 256, 0, stream>>>(deg, dinv, N);

    dim3 blk(256);
    dim3 g64 (cdiv(N, BM), 1);
    dim3 g640(cdiv(N, BM), 10);
    int nb_elem = cdiv(N * 64, 256);
    int nb_edge = cdiv(E * 32, 256);

    // layer 1: h1 = x @ Wb -> bufA ; xb = relu(P(h1) + bb) -> bufB
    k_gemm_wmma<<<g64, blk, 0, stream>>>(x, Wb_bf, nullptr, bufA, N, 64, 640, 0);
    k_self    <<<nb_elem, 256, 0, stream>>>(bufA, dinv, bufB, N);
    k_scatter <<<nb_edge, 256, 0, stream>>>(bufA, edges, dinv, bufB, E);
    k_bias_relu64<<<nb_elem, 256, 0, stream>>>(bufB, bb, N);

    // shared propagation: pxb = P(xb) -> bufC   (used by BOTH branches)
    k_self    <<<nb_elem, 256, 0, stream>>>(bufB, dinv, bufC, N);
    k_scatter <<<nb_edge, 256, 0, stream>>>(bufB, edges, dinv, bufC, E);

    // xs1 = relu(pxb @ Ws1 + bs1) -> bufA ; xw1 = relu(pxb @ Ww1 + bw1) -> bufB
    k_gemm_wmma<<<g64, blk, 0, stream>>>(bufC, Ws1_bf, bs1, bufA, N, 64, 64, 1);
    k_gemm_wmma<<<g64, blk, 0, stream>>>(bufC, Ww1_bf, bw1, bufB, N, 64, 64, 1);

    // pxs = P(xs1) -> bufC ; pxw = P(xw1) -> bufD   (one fused edge pass for both)
    k_self    <<<nb_elem, 256, 0, stream>>>(bufA, dinv, bufC, N);
    k_self    <<<nb_elem, 256, 0, stream>>>(bufB, dinv, bufD, N);
    k_scatter2<<<nb_edge, 256, 0, stream>>>(bufA, bufB, edges, dinv, bufC, bufD, E);

    // outputs: xs = relu(pxs @ Ws2 + bs2), xw = relu(pxw @ Ww2 + bw2)
    k_gemm_wmma<<<g640, blk, 0, stream>>>(bufC, Ws2_bf, bs2, out,                    N, 640, 64, 1);
    k_gemm_wmma<<<g640, blk, 0, stream>>>(bufD, Ww2_bf, bw2, out + (size_t)N * 640, N, 640, 64, 1);
}
